// LureSystem_34428457844831
// MI455X (gfx1250) — matrix operand order repaired and checked by reading, not compiled, not run on hardware
//
#include <hip/hip_runtime.h>

#define NXX 256
#define NDD 64
#define NEE 64
#define NWW 256
#define NBATCH 256
#define NSTEP 2048

typedef __bf16 v16bf __attribute__((ext_vector_type(16)));
typedef float v8f __attribute__((ext_vector_type(8)));
typedef float v4f __attribute__((ext_vector_type(4)));   // clang vector: valid for nontemporal builtins
typedef unsigned short u16t;

// LDS layout (units: u16)
#define XSTRIDE 264                  // 256 + 8 u16 pad (16B) -> conflict-free A-frag loads
#define C2F_OFF 0                    // C2 in B-fragment layout: 128 frags * 512 u16
#define AF_OFF  65536                // A  in B-fragment layout
#define X_OFF   131072               // x tiles, 2 buffers of [16][264]
#define XBUF    (16 * XSTRIDE)       // 4224 u16
#define W_OFF   (X_OFF + 2 * XBUF)   // w tiles, 2 buffers
#define ESCR_OFF (W_OFF + 2 * XBUF)  // e partial-sum scratch: 4 tiles * 256 f32 = 2048 u16*2
#define SMEM_U16 (ESCR_OFF + 2048 * 2) // 152064 u16 = 304128 bytes ( <= 320KB )

union Frag {
    v16bf v;
    u16t  h[16];
    uint4 q[2];
};

__device__ __forceinline__ u16t f2bf(float x) {
    union { float f; unsigned int u; } c;
    c.f = x;
    unsigned int r = c.u + 0x7FFFu + ((c.u >> 16) & 1u);
    return (u16t)(r >> 16);
}

// tanh(x) = sign(x) * (1 - e^{-2|x|}) / (1 + e^{-2|x|})
// exp2/rcp run on the TRANS pipe (co-executes with WMMA). ~5 ops, no overflow.
__device__ __forceinline__ float fast_tanh(float x) {
    float ax = fabsf(x);
    float t  = __builtin_amdgcn_exp2f(ax * -2.8853900817779268f);  // e^{-2|x|}
    float r  = (1.0f - t) * __builtin_amdgcn_rcpf(1.0f + t);
    return copysignf(r, x);
}

__device__ __forceinline__ v8f bfwmma(Frag a, Frag b, v8f c) {
    return __builtin_amdgcn_wmma_f32_16x16x32_bf16(false, a.v, false, b.v,
                                                   (short)0, c, false, false);
}

// A-operand fragment (16x32 bf16) from a row-major LDS tile [16][XSTRIDE].
__device__ __forceinline__ Frag ldsFragA(const u16t* tile, int lane, int k0) {
    const u16t* p = tile + (lane & 15) * XSTRIDE + k0 + ((lane >> 4) << 3);
    Frag f;
    f.q[0] = *(const uint4*)p;
    f.q[1] = *(const uint4*)(p + 16);
    return f;
}

// B-operand fragment pre-swizzled in LDS: 512 u16/frag, coalesced b128 loads.
__device__ __forceinline__ Frag ldsFragB(const u16t* fragBase, int lane) {
    Frag f;
    f.q[0] = *(const uint4*)(fragBase + lane * 8);
    f.q[1] = *(const uint4*)(fragBase + 256 + lane * 8);
    return f;
}

// B-operand fragment from row-major f32 weight W[n][k] in global memory (init only).
__device__ __forceinline__ Frag gFragB(const float* W, int ldw, int nBase, int kBase, int lane) {
    Frag f;
#pragma unroll
    for (int j = 0; j < 16; ++j)
        f.h[j] = f2bf(W[(size_t)(nBase + (lane & 15)) * ldw + kBase + (j & 7) + ((j >> 3) << 4) + ((lane >> 4) << 3)]);
    return f;
}

// A-operand fragment of the d tile from global with nontemporal b128 loads.
__device__ __forceinline__ Frag gFragDseq(const float* dseq, int b0, int k, int lane, int k0) {
    const float* p = dseq + (size_t)(b0 + (lane & 15)) * ((size_t)NSTEP * NDD)
                     + (size_t)k * NDD + k0 + ((lane >> 4) << 3);
    v4f lo  = __builtin_nontemporal_load((const v4f*)p);
    v4f hi  = __builtin_nontemporal_load((const v4f*)(p + 4));
    v4f lo2 = __builtin_nontemporal_load((const v4f*)(p + 16));
    v4f hi2 = __builtin_nontemporal_load((const v4f*)(p + 20));
    Frag f;
#pragma unroll
    for (int i = 0; i < 4; ++i) {
        f.h[i]      = f2bf(lo[i]);
        f.h[4 + i]  = f2bf(hi[i]);
        f.h[8 + i]  = f2bf(lo2[i]);
        f.h[12 + i] = f2bf(hi2[i]);
    }
    return f;
}

__global__ __attribute__((amdgpu_flat_work_group_size(256, 256)))
void lure_scan_kernel(const float* __restrict__ dseq, const float* __restrict__ x0,
                      const float* __restrict__ A,   const float* __restrict__ B,
                      const float* __restrict__ B2,  const float* __restrict__ C,
                      const float* __restrict__ Dm,  const float* __restrict__ D12,
                      const float* __restrict__ C2,  const float* __restrict__ D21,
                      float* __restrict__ out) {
    __shared__ u16t smem[SMEM_U16];
    float* escr = (float*)(smem + ESCR_OFF);   // [4 tiles][256] f32 partial sums for e

    const int tid  = threadIdx.x;
    const int lane = tid & 31;
    const int wv   = tid >> 5;            // 0..7
    const int b0   = blockIdx.x * 16;     // batch-row base of this workgroup
    const int et   = wv >> 1;             // e-tile shared by wave pair (0..3)

    // ---- one-time init: big weights into LDS in B-fragment layout (f32 -> bf16) ----
    for (int idx = tid; idx < 65536; idx += 256) {
        int frag   = idx >> 9;            // 0..127  (nTile*8 + kf)
        int within = idx & 511;
        int blk    = within >> 8;         // 0/1 -> k 8h+0..7 / 16+8h+0..7
        int l      = (within >> 3) & 31;  // lane (n = l & 15, h = l >> 4)
        int j      = within & 7;
        int nTile  = frag >> 3, kf = frag & 7;
        int n  = nTile * 16 + (l & 15);
        int kk = kf * 32 + blk * 16 + ((l >> 4) << 3) + j;
        int dst = frag * 512 + blk * 256 + l * 8 + j;
        smem[C2F_OFF + dst] = f2bf(C2[n * NXX + kk]);
        smem[AF_OFF  + dst] = f2bf(A[n * NXX + kk]);
    }
    // initial state x0 slice into x buffer 0 (row-major tile)
    for (int idx = tid; idx < 16 * 256; idx += 256) {
        int r = idx >> 8, c = idx & 255;
        smem[X_OFF + r * XSTRIDE + c] = f2bf(x0[(size_t)(b0 + r) * NXX + c]);
    }

    // ---- register-resident weight fragments (constant across all 2048 steps) ----
    const int t0 = wv * 2;                 // first owned N-tile of 16
    Frag B2f[2][8], Bf[2][2], D21f[2][2];
#pragma unroll
    for (int tt = 0; tt < 2; ++tt) {
#pragma unroll
        for (int kf = 0; kf < 8; ++kf)
            B2f[tt][kf] = gFragB(B2, NWW, (t0 + tt) * 16, kf * 32, lane);
#pragma unroll
        for (int kf = 0; kf < 2; ++kf) {
            Bf[tt][kf]   = gFragB(B,   NDD, (t0 + tt) * 16, kf * 32, lane);
            D21f[tt][kf] = gFragB(D21, NDD, (t0 + tt) * 16, kf * 32, lane);
        }
    }
    // e weights split across a wave pair: even wave holds C+D, odd wave holds D12.
    Frag Cf[8], Df[2], D12f[8];
    if (wv & 1) {
#pragma unroll
        for (int kf = 0; kf < 8; ++kf)
            D12f[kf] = gFragB(D12, NWW, et * 16, kf * 32, lane);
    } else {
#pragma unroll
        for (int kf = 0; kf < 8; ++kf)
            Cf[kf] = gFragB(C, NXX, et * 16, kf * 32, lane);
#pragma unroll
        for (int kf = 0; kf < 2; ++kf)
            Df[kf] = gFragB(Dm, NDD, et * 16, kf * 32, lane);
    }

    __syncthreads();

    const int   mAdd = (lane >> 4) << 3;   // accumulator: lanes 16-31 hold M = v+8
    const int   nLoc = lane & 15;
    const size_t XOUT = (size_t)NBATCH * NSTEP * NEE;
    const size_t WOUT = XOUT + (size_t)NBATCH * NXX;

    for (int k = 0; k < NSTEP; ++k) {
        const u16t* xcur = smem + X_OFF + (size_t)(k & 1) * XBUF;
        u16t*       xnxt = smem + X_OFF + (size_t)((k + 1) & 1) * XBUF;
        u16t*       wcur = smem + W_OFF + (size_t)(k & 1) * XBUF;

        // d_k A-fragments (every wave needs the full 16x64 tile; WGP$ absorbs reuse)
        Frag dF[2];
        dF[0] = gFragDseq(dseq, b0, k, lane, 0);
        dF[1] = gFragDseq(dseq, b0, k, lane, 32);

        // ---- fused compute: both products of x are taken in one pass over xf ----
        //   wa = x @ C2^T   (pre-tanh w)       xa = x @ A^T  (x_new partial)
        // 4 WMMAs per xf load, 4 independent accumulation chains.
        v8f wa[2], xa[2];
#pragma unroll
        for (int tt = 0; tt < 2; ++tt)
#pragma unroll
            for (int i = 0; i < 8; ++i) { wa[tt][i] = 0.f; xa[tt][i] = 0.f; }

#pragma unroll
        for (int kf = 0; kf < 8; ++kf) {
            Frag xf = ldsFragA(xcur, lane, kf * 32);
            Frag c0 = ldsFragB(smem + C2F_OFF + ((t0    ) * 8 + kf) * 512, lane);
            Frag c1 = ldsFragB(smem + C2F_OFF + ((t0 + 1) * 8 + kf) * 512, lane);
            Frag a0 = ldsFragB(smem + AF_OFF  + ((t0    ) * 8 + kf) * 512, lane);
            Frag a1 = ldsFragB(smem + AF_OFF  + ((t0 + 1) * 8 + kf) * 512, lane);
            wa[0] = bfwmma(xf, c0, wa[0]);
            wa[1] = bfwmma(xf, c1, wa[1]);
            xa[0] = bfwmma(xf, a0, xa[0]);
            xa[1] = bfwmma(xf, a1, xa[1]);
        }
#pragma unroll
        for (int tt = 0; tt < 2; ++tt) {
            wa[tt] = bfwmma(dF[0], D21f[tt][0], wa[tt]);   // + d @ D21^T
            wa[tt] = bfwmma(dF[1], D21f[tt][1], wa[tt]);
            xa[tt] = bfwmma(dF[0], Bf[tt][0], xa[tt]);     // + d @ B^T
            xa[tt] = bfwmma(dF[1], Bf[tt][1], xa[tt]);
        }

        float tw[2][8];
#pragma unroll
        for (int tt = 0; tt < 2; ++tt)
#pragma unroll
            for (int v = 0; v < 8; ++v)
                tw[tt][v] = fast_tanh(wa[tt][v]);
#pragma unroll
        for (int tt = 0; tt < 2; ++tt) {
            int nc = (t0 + tt) * 16 + nLoc;
#pragma unroll
            for (int v = 0; v < 8; ++v)
                wcur[(v + mAdd) * XSTRIDE + nc] = f2bf(tw[tt][v]);
        }
        if (k == NSTEP - 1) {              // w_out = last w (f32), uniform branch
#pragma unroll
            for (int tt = 0; tt < 2; ++tt)
#pragma unroll
                for (int v = 0; v < 8; ++v)
                    out[WOUT + (size_t)(b0 + v + mAdd) * NWW + (t0 + tt) * 16 + nLoc] = tw[tt][v];
        }
        __syncthreads();   // barrier 1: w tile complete

        // ---- phase 2: x_new += w @ B2^T ; odd waves also build e partial w @ D12^T
        //      (wf fragments shared by all products -> zero extra LDS traffic) ----
        if (wv & 1) {
            v8f pa;
#pragma unroll
            for (int i = 0; i < 8; ++i) pa[i] = 0.f;
#pragma unroll
            for (int kf = 0; kf < 8; ++kf) {
                Frag wf = ldsFragA(wcur, lane, kf * 32);
                xa[0] = bfwmma(wf, B2f[0][kf], xa[0]);
                xa[1] = bfwmma(wf, B2f[1][kf], xa[1]);
                pa    = bfwmma(wf, D12f[kf], pa);
            }
#pragma unroll
            for (int v = 0; v < 8; ++v)
                escr[et * 256 + (v + mAdd) * 16 + nLoc] = pa[v];
        } else {
#pragma unroll
            for (int kf = 0; kf < 8; ++kf) {
                Frag wf = ldsFragA(wcur, lane, kf * 32);
                xa[0] = bfwmma(wf, B2f[0][kf], xa[0]);
                xa[1] = bfwmma(wf, B2f[1][kf], xa[1]);
            }
        }
#pragma unroll
        for (int tt = 0; tt < 2; ++tt) {
            int nc = (t0 + tt) * 16 + nLoc;
#pragma unroll
            for (int v = 0; v < 8; ++v)
                xnxt[(v + mAdd) * XSTRIDE + nc] = f2bf(xa[tt][v]);
        }
        if (k == NSTEP - 1) {              // x_out = final state (f32), uniform branch
#pragma unroll
            for (int tt = 0; tt < 2; ++tt)
#pragma unroll
                for (int v = 0; v < 8; ++v)
                    out[XOUT + (size_t)(b0 + v + mAdd) * NXX + (t0 + tt) * 16 + nLoc] = xa[tt][v];
        }
        __syncthreads();   // barrier 2: x_new tile + e partial complete

        // ---- phase 3 (even waves): e = x_new @ C^T + d @ D^T + partner partial ----
        if ((wv & 1) == 0) {
            v8f ea;
#pragma unroll
            for (int i = 0; i < 8; ++i) ea[i] = 0.f;
#pragma unroll
            for (int kf = 0; kf < 8; ++kf) {
                Frag xf = ldsFragA(xnxt, lane, kf * 32);
                ea = bfwmma(xf, Cf[kf], ea);
            }
            ea = bfwmma(dF[0], Df[0], ea);
            ea = bfwmma(dF[1], Df[1], ea);

            int nc = et * 16 + nLoc;
#pragma unroll
            for (int v = 0; v < 8; ++v) {
                int m = v + mAdd;
                float val = ea[v] + escr[et * 256 + m * 16 + nLoc];
                // e_hat streamed once: nontemporal store keeps L2 for weights/state
                __builtin_nontemporal_store(val,
                    out + ((size_t)(b0 + m) * NSTEP + k) * NEE + nc);
            }
        }
        // next step writes the other x/w buffers; the two barriers order buffer reuse.
        // escr reuse is safe: odd writes it after barrier 1 of step k+1, which even
        // waves only release after finishing this phase-3 read.
    }
}

extern "C" void kernel_launch(void* const* d_in, const int* in_sizes, int n_in,
                              void* d_out, int out_size, void* d_ws, size_t ws_size,
                              hipStream_t stream) {
    (void)in_sizes; (void)n_in; (void)out_size; (void)d_ws; (void)ws_size;
    const float* dseq = (const float*)d_in[0];
    const float* x0   = (const float*)d_in[1];
    const float* A    = (const float*)d_in[2];
    const float* B    = (const float*)d_in[3];
    const float* B2   = (const float*)d_in[4];
    const float* C    = (const float*)d_in[5];
    const float* Dm   = (const float*)d_in[6];
    const float* D12  = (const float*)d_in[7];
    const float* C2   = (const float*)d_in[8];
    const float* D21  = (const float*)d_in[9];
    float* out = (float*)d_out;

    lure_scan_kernel<<<NBATCH / 16, 256, 0, stream>>>(dseq, x0, A, B, B2, C, Dm,
                                                      D12, C2, D21, out);
}
